// TalkingHeadAttention_85100482003670
// MI455X (gfx1250) — matrix operand order repaired and checked by reading, not compiled, not run on hardware
//
#include <hip/hip_runtime.h>

// ---------------------------------------------------------------------------
// Talking-heads attention for MI455X (gfx1250), wave32 + WMMA bf16.
// B=2, N=2048, D=1024, H=16, HD=64.
// ---------------------------------------------------------------------------

#define BATCH 2
#define SEQ   2048
#define DIM   1024
#define HEADS 16
#define HDIM  64
#define MROWS (BATCH * SEQ)   // 4096

typedef __attribute__((ext_vector_type(16))) __bf16 v16bf;
typedef __attribute__((ext_vector_type(8)))  __bf16 v8bf;
typedef __attribute__((ext_vector_type(8)))  float  v8f;

// ---- bf16 helpers (bit-exact RNE convert)
static __device__ __forceinline__ __bf16 f2bf(float f) {
    union { float f; unsigned u; } c; c.f = f;
    unsigned u = c.u + 0x7FFFu + ((c.u >> 16) & 1u);
    unsigned short h = (unsigned short)(u >> 16);
    __bf16 b; __builtin_memcpy(&b, &h, 2); return b;
}
static __device__ __forceinline__ float bf2f(__bf16 b) {
    unsigned short h; __builtin_memcpy(&h, &b, 2);
    union { float f; unsigned u; } c; c.u = ((unsigned)h) << 16; return c.f;
}
static __device__ __forceinline__ v16bf cat8(v8bf lo, v8bf hi) {
    return __builtin_shufflevector(lo, hi, 0,1,2,3,4,5,6,7,8,9,10,11,12,13,14,15);
}

// WMMA 16x16x32 bf16 fragment maps (CDNA5 ISA 7.12.2):
// A: lane = half*16 + r holds row r; this code uses elem j -> K = j + 16*half (same order as B)
//    (two 8-element K runs: [16*half, 16*half+8) and [16*half+8, 16*half+16))
// B: lane = half*16 + r holds col r; this code uses elem j -> K = j + 16*half (16 contiguous K)
// C: elem i -> row = i + 8*half, col = r

// ===========================================================================
// Kernel 0a: f32 -> bf16 convert (row-major passthrough)
// ===========================================================================
__global__ void cvt_f32_bf16(const float* __restrict__ in, __bf16* __restrict__ out, int n) {
    int i = blockIdx.x * blockDim.x + threadIdx.x;
    int stride = gridDim.x * blockDim.x;
    for (; i < n; i += stride) out[i] = f2bf(in[i]);
}

// ===========================================================================
// Kernel 0b: f32 [K][N] -> bf16 transposed [N][K] (tiled through LDS)
// ===========================================================================
__global__ __launch_bounds__(256) void cvt_transpose_bf16(
    const float* __restrict__ in, __bf16* __restrict__ out, int K, int N) {
    __shared__ float t[32][33];
    const int tx = threadIdx.x & 31;
    const int ty = threadIdx.x >> 5;            // 0..7
    const int kb = blockIdx.y * 32, nb = blockIdx.x * 32;
    #pragma unroll
    for (int i = 0; i < 4; ++i) {
        int kk = ty + i * 8;
        t[kk][tx] = in[(size_t)(kb + kk) * N + nb + tx];
    }
    __syncthreads();
    #pragma unroll
    for (int i = 0; i < 4; ++i) {
        int nn = ty + i * 8;
        out[(size_t)(nb + nn) * K + kb + tx] = f2bf(t[tx][nn]);
    }
}

// ===========================================================================
// Kernel 1: tiled bf16 WMMA GEMM.  C[M x N] = scale * (A[M x K] @ Wt^T)
// A row-major [M][K]; Wt is the TRANSPOSED weight, row-major [N][K].
// BM=128 BN=64 BK=32, 8 waves. LDS staged in fragment-friendly layouts so
// every WMMA operand load is 2x ds_load_b128.
// ===========================================================================
__global__ __launch_bounds__(256) void gemm_bf16_wmma(
    const __bf16* __restrict__ A, const __bf16* __restrict__ Wt,
    __bf16* __restrict__ Cbf, float* __restrict__ Cf,
    const float* __restrict__ bias, float scale, int M, int Nn, int K)
{
    // A staged fragment-major: [sub 0..7][lane 0..31][16 elems]
    __shared__ __bf16 sA[8 * 32 * 16];   // 4096 elems, 8KB
    // B staged as Wt rows: [col 0..63][32 k]
    __shared__ __bf16 sB[64 * 32];       // 2048 elems, 4KB

    const int tid  = threadIdx.x;
    const int w    = tid >> 5;
    const int lane = tid & 31;
    const int r    = lane & 15;
    const int half = lane >> 4;
    const int rowBase = blockIdx.y * 128;
    const int colBase = blockIdx.x * 64;

    v8f acc[4] = {};

    for (int k0 = 0; k0 < K; k0 += 32) {
        __syncthreads();
        // A tile: 512 16-byte chunks; chunk = (row, c) with c = which 8-K run
        #pragma unroll
        for (int ci = tid; ci < 512; ci += 256) {
            int row = ci >> 2, c = ci & 3;
            v8bf val = *(const v8bf*)(A + (size_t)(rowBase + row) * K + k0 + c * 8);
            // dest lane = (c>>1)*16 + (row&15); j-run = (c&1)*8 ; sub = row>>4
            int dst = (((row >> 4) * 32 + (c >> 1) * 16 + (row & 15)) << 4) + ((c & 1) * 8);
            *(v8bf*)(sA + dst) = val;
        }
        // B tile: 256 16-byte chunks (one per thread)
        {
            int n = tid >> 2, c = tid & 3;
            v8bf val = *(const v8bf*)(Wt + (size_t)(colBase + n) * K + k0 + c * 8);
            *(v8bf*)(sB + n * 32 + c * 8) = val;
        }
        __syncthreads();

        v16bf af = *(const v16bf*)(sA + ((w * 32 + lane) << 4));
        #pragma unroll
        for (int t = 0; t < 4; ++t) {
            v16bf bf_ = *(const v16bf*)(sB + (t * 16 + r) * 32 + 16 * half);
            acc[t] = __builtin_amdgcn_wmma_f32_16x16x32_bf16(
                false, af, false, bf_, (short)0, acc[t], false, false);
        }
    }

    #pragma unroll
    for (int t = 0; t < 4; ++t) {
        #pragma unroll
        for (int i = 0; i < 8; ++i) {
            int row = rowBase + w * 16 + i + 8 * half;
            int col = colBase + t * 16 + r;
            float vvl = acc[t][i] * scale;
            if (Cbf) Cbf[(size_t)row * Nn + col] = f2bf(vvl);
            else     Cf [(size_t)row * Nn + col] = vvl + (bias ? bias[col] : 0.f);
        }
    }
}

// ===========================================================================
// Kernel 2: fused talking-heads attention.
// Block = 512 threads = 16 waves (wave g owns head g). q-tile = 32 rows.
// Pass A -> softmax stats of L-mixed scores; pass B -> normalized probs,
// Wm mix, P~ @ V via WMMA.  Dynamic LDS 256KB.
// ===========================================================================
__global__ __launch_bounds__(512) void attn_talking_heads(
    const __bf16* __restrict__ q, const __bf16* __restrict__ k,
    const __bf16* __restrict__ v, const float* __restrict__ Lmat,
    const float* __restrict__ Wmmat, __bf16* __restrict__ out)
{
    extern __shared__ char smem[];
    __bf16* kbuf  = (__bf16*)smem;                      // 32 x 1024
    __bf16* vbuf  = kbuf + 32 * 1024;                   // 32 x 1024
    float*  sbuf  = (float*)(vbuf + 32 * 1024);         // [16][32][32] f32
    __bf16* pbuf  = (__bf16*)(sbuf + 16 * 32 * 32);     // [16][32][32] bf16
    __bf16* ptbuf = pbuf + 16 * 32 * 32;                // [16][32][32] bf16

    const int tid   = threadIdx.x;
    const int g     = tid >> 5;          // wave == head
    const int lane  = tid & 31;
    const int r     = lane & 15;
    const int half  = lane >> 4;
    const int b     = blockIdx.x >> 6;   // 64 q-tiles per batch
    const int qbase = (blockIdx.x & 63) * 32;

    float Lg[16], Wg[16];
    #pragma unroll
    for (int h = 0; h < 16; ++h) { Lg[h] = Lmat[g * 16 + h]; Wg[h] = Wmmat[g * 16 + h]; }

    // Q fragments: two aligned 16B global loads per fragment
    v16bf aq[2][2];
    #pragma unroll
    for (int rh = 0; rh < 2; ++rh)
        #pragma unroll
        for (int kc = 0; kc < 2; ++kc) {
            const __bf16* base = q + (size_t)(b * SEQ + qbase + rh * 16 + r) * DIM
                                   + g * HDIM + kc * 32;
            v8bf lo = *(const v8bf*)(base + 16 * half);
            v8bf hi = *(const v8bf*)(base + 16 * half + 8);
            aq[rh][kc] = cat8(lo, hi);
        }

    float m_run = -1e30f, z_run = 0.f;   // lane L holds stats of q-row L

    // ---------------- Pass A: softmax stats of L-mixed scores ----------------
    for (int mt = 0; mt < SEQ / 32; ++mt) {
        const int mbase = mt * 32;
        __syncthreads();
        // K tile: 4096 16-byte chunks, 512 threads -> 8 each
        for (int ci = tid; ci < 4096; ci += 512) {
            int rr = ci >> 7, cc = (ci & 127) * 8;
            *(v8bf*)(kbuf + rr * 1024 + cc) =
                *(const v8bf*)(k + (size_t)(b * SEQ + mbase + rr) * DIM + cc);
        }
        __syncthreads();

        // S_g = Q_g @ K_g^T (32x32) -> exchange buffer
        #pragma unroll
        for (int rh = 0; rh < 2; ++rh) {
            #pragma unroll
            for (int ch = 0; ch < 2; ++ch) {
                v8f acc = {};
                #pragma unroll
                for (int kc = 0; kc < 2; ++kc) {
                    v16bf bf_ = *(const v16bf*)(kbuf + (ch * 16 + r) * 1024
                                                + g * HDIM + kc * 32 + 16 * half);
                    acc = __builtin_amdgcn_wmma_f32_16x16x32_bf16(
                        false, aq[rh][kc], false, bf_, (short)0, acc, false, false);
                }
                #pragma unroll
                for (int i = 0; i < 8; ++i)
                    sbuf[(g * 32 + rh * 16 + i + 8 * half) * 32 + ch * 16 + r] = acc[i];
            }
        }
        __syncthreads();

        // L head-mix + online (max, Z) update; lane=column, loop rows
        for (int row = 0; row < 32; ++row) {
            float sp = 0.f;
            #pragma unroll
            for (int h = 0; h < 16; ++h)
                sp = fmaf(Lg[h], sbuf[(h * 32 + row) * 32 + lane], sp);
            float mx = sp;
            for (int off = 16; off > 0; off >>= 1) mx = fmaxf(mx, __shfl_xor(mx, off, 32));
            float mold = __shfl(m_run, row, 32);
            float zold = __shfl(z_run, row, 32);
            float mnew = fmaxf(mold, mx);
            float e = __expf(sp - mnew);
            float se = e;
            for (int off = 16; off > 0; off >>= 1) se += __shfl_xor(se, off, 32);
            float znew = zold * __expf(mold - mnew) + se;
            if (lane == row) { m_run = mnew; z_run = znew; }
        }
    }

    // ---------------- Pass B: normalize, Wm mix, accumulate P~ @ V ----------
    v8f oacc[2][4] = {};
    for (int mt = 0; mt < SEQ / 32; ++mt) {
        const int mbase = mt * 32;
        __syncthreads();
        for (int ci = tid; ci < 4096; ci += 512) {
            int rr = ci >> 7, cc = (ci & 127) * 8;
            size_t gidx = (size_t)(b * SEQ + mbase + rr) * DIM + cc;
            *(v8bf*)(kbuf + rr * 1024 + cc) = *(const v8bf*)(k + gidx);
            *(v8bf*)(vbuf + rr * 1024 + cc) = *(const v8bf*)(v + gidx);
        }
        __syncthreads();

        // recompute raw scores (flash-2 style)
        #pragma unroll
        for (int rh = 0; rh < 2; ++rh) {
            #pragma unroll
            for (int ch = 0; ch < 2; ++ch) {
                v8f acc = {};
                #pragma unroll
                for (int kc = 0; kc < 2; ++kc) {
                    v16bf bf_ = *(const v16bf*)(kbuf + (ch * 16 + r) * 1024
                                                + g * HDIM + kc * 32 + 16 * half);
                    acc = __builtin_amdgcn_wmma_f32_16x16x32_bf16(
                        false, aq[rh][kc], false, bf_, (short)0, acc, false, false);
                }
                #pragma unroll
                for (int i = 0; i < 8; ++i)
                    sbuf[(g * 32 + rh * 16 + i + 8 * half) * 32 + ch * 16 + r] = acc[i];
            }
        }
        __syncthreads();

        // p_g = exp(L-mix - m)/Z  -> pbuf (bf16)
        for (int row = 0; row < 32; ++row) {
            float sp = 0.f;
            #pragma unroll
            for (int h = 0; h < 16; ++h)
                sp = fmaf(Lg[h], sbuf[(h * 32 + row) * 32 + lane], sp);
            float mrow = __shfl(m_run, row, 32);
            float zrow = __shfl(z_run, row, 32);
            pbuf[(g * 32 + row) * 32 + lane] = f2bf(__expf(sp - mrow) / zrow);
        }
        __syncthreads();

        // p~_g = sum_h Wm[g,h] * p_h  -> ptbuf (bf16)
        for (int row = 0; row < 32; ++row) {
            float pt = 0.f;
            #pragma unroll
            for (int h = 0; h < 16; ++h)
                pt = fmaf(Wg[h], bf2f(pbuf[(h * 32 + row) * 32 + lane]), pt);
            ptbuf[(g * 32 + row) * 32 + lane] = f2bf(pt);
        }
        __syncthreads();

        // V B-fragments (strided K=m; hoisted, shared across row halves)
        v16bf vF[4];
        #pragma unroll
        for (int ct = 0; ct < 4; ++ct)
            #pragma unroll
            for (int j = 0; j < 16; ++j)
                vF[ct][j] = vbuf[(j + 16 * half) * 1024 + g * HDIM + ct * 16 + r];

        // O_g += p~_g[32x32] @ V_g[32x64]
        #pragma unroll
        for (int rh = 0; rh < 2; ++rh) {
            const __bf16* pb = ptbuf + (g * 32 + rh * 16 + r) * 32;
            v16bf af = cat8(*(const v8bf*)(pb + 16 * half),
                            *(const v8bf*)(pb + 16 * half + 8));
            #pragma unroll
            for (int ct = 0; ct < 4; ++ct)
                oacc[rh][ct] = __builtin_amdgcn_wmma_f32_16x16x32_bf16(
                    false, af, false, vF[ct], (short)0, oacc[rh][ct], false, false);
        }
    }

    // write O (bf16) for the output projection GEMM
    #pragma unroll
    for (int rh = 0; rh < 2; ++rh)
        #pragma unroll
        for (int ct = 0; ct < 4; ++ct)
            #pragma unroll
            for (int i = 0; i < 8; ++i) {
                int row = qbase + rh * 16 + i + 8 * half;
                int col = g * HDIM + ct * 16 + r;
                out[(size_t)(b * SEQ + row) * DIM + col] = f2bf(oacc[rh][ct][i]);
            }
}

// ===========================================================================
// Host launcher
// ===========================================================================
extern "C" void kernel_launch(void* const* d_in, const int* in_sizes, int n_in,
                              void* d_out, int out_size, void* d_ws, size_t ws_size,
                              hipStream_t stream) {
    const float* x  = (const float*)d_in[0];
    const float* Wq = (const float*)d_in[1];
    const float* Wk = (const float*)d_in[2];
    const float* Wv = (const float*)d_in[3];
    const float* L  = (const float*)d_in[4];
    const float* Wm = (const float*)d_in[5];
    const float* Wo = (const float*)d_in[6];
    const float* bo = (const float*)d_in[7];
    float* out = (float*)d_out;

    // workspace layout (bf16); weights stored TRANSPOSED [N][K]
    __bf16* xbf  = (__bf16*)d_ws;
    __bf16* wqT  = xbf + (size_t)MROWS * DIM;
    __bf16* wkT  = wqT + (size_t)DIM * DIM;
    __bf16* wvT  = wkT + (size_t)DIM * DIM;
    __bf16* woT  = wvT + (size_t)DIM * DIM;
    __bf16* qb   = woT + (size_t)DIM * DIM;
    __bf16* kb   = qb  + (size_t)MROWS * DIM;
    __bf16* vb   = kb  + (size_t)MROWS * DIM;
    __bf16* aob  = vb  + (size_t)MROWS * DIM;

    // 1) convert x (row-major) and weights (transposed) to bf16
    cvt_f32_bf16<<<4096, 256, 0, stream>>>(x, xbf, MROWS * DIM);
    dim3 tg(DIM / 32, DIM / 32);
    cvt_transpose_bf16<<<tg, 256, 0, stream>>>(Wq, wqT, DIM, DIM);
    cvt_transpose_bf16<<<tg, 256, 0, stream>>>(Wk, wkT, DIM, DIM);
    cvt_transpose_bf16<<<tg, 256, 0, stream>>>(Wv, wvT, DIM, DIM);
    cvt_transpose_bf16<<<tg, 256, 0, stream>>>(Wo, woT, DIM, DIM);

    // 2) QKV projections (K pre-scaled by HD^-0.5 = 0.125)
    dim3 gg(DIM / 64, MROWS / 128);
    gemm_bf16_wmma<<<gg, 256, 0, stream>>>(xbf, wqT, qb, nullptr, nullptr, 1.0f,   MROWS, DIM, DIM);
    gemm_bf16_wmma<<<gg, 256, 0, stream>>>(xbf, wkT, kb, nullptr, nullptr, 0.125f, MROWS, DIM, DIM);
    gemm_bf16_wmma<<<gg, 256, 0, stream>>>(xbf, wvT, vb, nullptr, nullptr, 1.0f,   MROWS, DIM, DIM);

    // 3) fused talking-heads attention (256KB dynamic LDS per block)
    attn_talking_heads<<<BATCH * (SEQ / 32), 512, 256 * 1024, stream>>>(qb, kb, vb, L, Wm, aob);

    // 4) output projection + bias (f32 out)
    gemm_bf16_wmma<<<gg, 256, 0, stream>>>(aob, woT, nullptr, out, bo, 1.0f, MROWS, DIM, DIM);
}
